// SpeakerEncoder_81003083203396
// MI455X (gfx1250) — matrix operand order repaired and checked by reading, not compiled, not run on hardware
//
#include <hip/hip_runtime.h>
#include <hip/hip_bf16.h>

// ---------------- problem constants ----------------
#define NN   64
#define MM   10
#define TT   160
#define INF_ 40
#define HH   256
#define BB   640            // N*M
#define ROWS (BB*TT)        // 102400 time-major rows
#define K0P  64             // layer0 input padded 40 -> 64
#define LDSX 264            // h_lds row stride (bf16): 528B, 16B aligned, bank-skewed

typedef __attribute__((ext_vector_type(16))) __bf16 v16bf;
typedef __attribute__((ext_vector_type(8)))  __bf16 v8bf;
typedef __attribute__((ext_vector_type(8)))  float  v8f;

// global-address-space view of a weight tile -> lowers to global_load_b128 (LOADcnt
// only) instead of flat_load_b128 (which also ties up DScnt; ISA sec 10.1.1)
typedef const v16bf __attribute__((address_space(1)))* gv16p;

union ABf16 { v16bf v; v8bf h[2]; };

__device__ __forceinline__ float sigmoidf_(float x) { return 1.0f / (1.0f + __expf(-x)); }

// ---------------- conversion kernels ----------------
__global__ void k_convert_seq(const float* __restrict__ seq, __bf16* __restrict__ x0) {
    int idx = blockIdx.x * blockDim.x + threadIdx.x;      // over ROWS*K0P
    if (idx >= ROWS * K0P) return;
    int k   = idx & (K0P - 1);
    int row = idx >> 6;                                    // t*B + b
    int t   = row / BB;
    int b   = row % BB;
    float v = (k < INF_) ? seq[((size_t)b * TT + t) * INF_ + k] : 0.0f;
    x0[idx] = (__bf16)v;
}

__global__ void k_convert_w(const float* __restrict__ W, __bf16* __restrict__ Wo,
                            int ldi, int ldo) {
    int idx = blockIdx.x * blockDim.x + threadIdx.x;      // over 1024*ldo
    if (idx >= 4 * HH * ldo) return;
    int c = idx % ldo, r = idx / ldo;
    Wo[idx] = (__bf16)((c < ldi) ? W[(size_t)r * ldi + c] : 0.0f);
}

__global__ void k_bias_sum(const float* __restrict__ bih, const float* __restrict__ bhh,
                           float* __restrict__ bsum) {
    int i = blockIdx.x * blockDim.x + threadIdx.x;
    if (i < 4 * HH) bsum[i] = bih[i] + bhh[i];
}

// ---------------- fused LSTM layer (recurrence + input projection) ----------------
// grid: 40 blocks (batch tiles of 16); block: 256 threads = 8 waves.
// Wave w owns hidden units [32w,32w+32): eight gate tiles (i/f/g/o x 2 cols) and the
// cell state c live in WMMA C/D-layout v8f registers for all 160 steps.
// The weight stream is loop-invariant across the 160 steps; opaque asm pointer
// rotations (zero instructions) stop LICM from hoisting+spilling it, and a fake acc
// dependency pins the modulo-3 distance-2 register pipeline of 4-tile half-groups
// (~8 WMMAs of L2-latency cover per wave, x2 waves/SIMD).
template <int KIN>
__global__ __launch_bounds__(256, 1)
void k_lstm_layer(const __bf16* __restrict__ xin,
                  const __bf16* __restrict__ Wi, const __bf16* __restrict__ Wh,
                  const float* __restrict__ bsum, const int* __restrict__ lens,
                  __bf16* __restrict__ hout, float* __restrict__ cout) {
    constexpr int KCX  = KIN / 32;        // x K-chunks (2 or 8)
    constexpr int G    = KCX + 8;         // K-groups per step (x then h)
    constexpr int HG   = 2 * G;           // 4-tile half-groups per step
    constexpr int XLDS = KIN + 8;         // x_lds row stride (bf16)

    __shared__ __bf16 h_lds[16 * LDSX];
    __shared__ __bf16 x_lds[2][16 * XLDS];

    const int tid    = threadIdx.x;
    const int wave   = tid >> 5;           // 0..7
    const int lane   = tid & 31;
    const int laneHi = lane >> 4;          // K-half selector (A & B layouts)
    const int nloc   = lane & 15;          // row (A) / column (B) within 16
    const int b0     = blockIdx.x * 16;
    const int hbase  = wave * 32;          // hidden-unit base of this wave (32 units)

    // weight base addresses as opaque-rotatable values (byte addresses)
    size_t wiA = (size_t)(Wi + (size_t)(hbase + nloc) * KIN + laneHi * 16);
    size_t whA = (size_t)(Wh + (size_t)(hbase + nloc) * HH  + laneHi * 16);
    // half-group hg covers gates {2*(hg&1), 2*(hg&1)+1} x column tiles {0,1}, K-group hg>>1
    auto wtile = [&](size_t wiA_, size_t whA_, int hg, int q) -> gv16p {
        int gi = hg >> 1, half = hg & 1;
        int g = 2 * half + (q >> 1), j = q & 1;
        return (gi < KCX)
            ? (gv16p)(wiA_ + 2 * ((size_t)(g * HH + 16 * j) * KIN + gi * 32))
            : (gv16p)(whA_ + 2 * ((size_t)(g * HH + 16 * j) * HH  + (gi - KCX) * 32));
    };
    auto aptr = [&](int gi, int xb) -> const __bf16* {
        return (gi < KCX) ? &x_lds[xb][nloc * XLDS + gi * 32 + laneHi * 16]
                          : &h_lds[nloc * LDSX + (gi - KCX) * 32 + laneHi * 16];
    };
    auto stage_x = [&](int t, int xb) {
        constexpr int xtot = 16 * KIN;
        for (int e = tid * 8; e < xtot; e += 256 * 8) {
            int r = e / KIN, cc = e % KIN;
            *(v8bf*)&x_lds[xb][r * XLDS + cc] =
                *(const v8bf*)&xin[((size_t)t * BB + b0 + r) * KIN + cc];
        }
    };

    float bias[4][2];
#pragma unroll
    for (int g = 0; g < 4; ++g)
#pragma unroll
        for (int j = 0; j < 2; ++j)
            bias[g][j] = bsum[g * HH + hbase + 16 * j + nloc];

    // sequence lengths for the 8 batch rows this lane's accumulator elements cover
    int len8[8];
#pragma unroll
    for (int i = 0; i < 8; ++i) len8[i] = lens[b0 + i + laneHi * 8];

    v8f c[2] = {}; v8f h[2] = {};

    // prologue: zero h state, stage x(0), prefetch half-groups 0,1
    for (int i = tid; i < 16 * LDSX; i += 256) h_lds[i] = (__bf16)0.0f;
    stage_x(0, 0);
    v16bf wb[3][4];
#pragma unroll
    for (int q = 0; q < 4; ++q) {
        wb[0][q] = *wtile(wiA, whA, 0, q);
        wb[1][q] = *wtile(wiA, whA, 1, q);
    }

#pragma unroll 1
    for (int t = 0; t < TT; ++t) {
        // opaque rotation: compiler must assume the bases changed -> no LICM of the
        // per-step weight loads (they are L2-hot; hoisting spills them to scratch)
        asm volatile("" : "+v"(wiA), "+v"(whA));

        const int xb = t & 1;
        __syncthreads();                   // x(t) staged, h_lds(t-1) written

        ABf16 ab[2];
        {   const __bf16* ap = aptr(0, xb);
            ab[0].h[0] = *(const v8bf*)ap; ab[0].h[1] = *(const v8bf*)(ap + 8); }

        v8f acc[4][2] = {};                // bias added in elementwise phase

        // ---- pipelined GEMM over all x- and h- half-groups ----
#pragma unroll
        for (int hg = 0; hg < HG; ++hg) {
            const int gi = hg >> 1, half = hg & 1;
            if (hg + 2 < HG) {             // prefetch weights 2 half-groups ahead;
                                           // fake acc dependency pins the loads here
                float dep = acc[0][0][0];
                asm volatile("" : "+v"(wiA), "+v"(whA) : "v"(dep));
#pragma unroll
                for (int q = 0; q < 4; ++q)
                    wb[(hg + 2) % 3][q] = *wtile(wiA, whA, hg + 2, q);
            }
            if (half == 0 && gi + 1 < G) { // prefetch A-tile 1 K-group ahead (LDS)
                const __bf16* ap = aptr(gi + 1, xb);
                ab[(gi + 1) & 1].h[0] = *(const v8bf*)ap;
                ab[(gi + 1) & 1].h[1] = *(const v8bf*)(ap + 8);
            }
#pragma unroll
            for (int q = 0; q < 4; ++q) {
                const int g = 2 * half + (q >> 1), j = q & 1;
                acc[g][j] = __builtin_amdgcn_wmma_f32_16x16x32_bf16(
                    false, ab[gi & 1].v, false, wb[hg % 3][q],
                    (short)0, acc[g][j], false, false);
            }
        }

        // prefetch next step's half-groups 0,1 (same addresses, L2-hot)
#pragma unroll
        for (int q = 0; q < 4; ++q) {
            wb[0][q] = *wtile(wiA, whA, 0, q);
            wb[1][q] = *wtile(wiA, whA, 1, q);
        }
        // stage x(t+1) into the alternate LDS buffer (overlaps with elementwise)
        if (t + 1 < TT) stage_x(t + 1, xb ^ 1);

        // ---- LSTM elementwise update (accumulator layout, fp32) ----
#pragma unroll
        for (int j = 0; j < 2; ++j)
#pragma unroll
            for (int i = 0; i < 8; ++i) {
                float ig = sigmoidf_(acc[0][j][i] + bias[0][j]);
                float fg = sigmoidf_(acc[1][j][i] + bias[1][j]);
                float gg = tanhf(acc[2][j][i] + bias[2][j]);
                float og = sigmoidf_(acc[3][j][i] + bias[3][j]);
                float cn = fg * c[j][i] + ig * gg;
                float hn = og * tanhf(cn);
                bool valid = t < len8[i];
                c[j][i] = valid ? cn : c[j][i];
                h[j][i] = valid ? hn : h[j][i];
            }

        __syncthreads();                   // all h_lds/x_lds reads of step t done
#pragma unroll
        for (int j = 0; j < 2; ++j)
#pragma unroll
            for (int i = 0; i < 8; ++i) {
                int r = i + laneHi * 8;
                __bf16 hb = (__bf16)h[j][i];
                h_lds[r * LDSX + hbase + 16 * j + nloc] = hb;
                hout[((size_t)t * BB + b0 + r) * HH + hbase + 16 * j + nloc] = hb;
            }
    }

    // c holds the cell state at each row's last valid step (mask held it)
#pragma unroll
    for (int j = 0; j < 2; ++j)
#pragma unroll
        for (int i = 0; i < 8; ++i) {
            int r = i + laneHi * 8;
            cout[(size_t)(b0 + r) * HH + hbase + 16 * j + nloc] = c[j][i];
        }
}

// ---------------- similarity head ----------------
__device__ __forceinline__ float blockReduceSum256(float v, float* red) {
    int t = threadIdx.x;
    __syncthreads();
    red[t] = v;
    __syncthreads();
    for (int s = 128; s > 0; s >>= 1) {
        if (t < s) red[t] += red[t + s];
        __syncthreads();
    }
    return red[0];
}

__global__ void k_embed_norm(const float* __restrict__ E, float* __restrict__ En,
                             float* __restrict__ Cn, float* __restrict__ Cmn) {
    __shared__ float red[256];
    int n = blockIdx.x, j = threadIdx.x;     // 64 blocks x 256 threads
    float ep[MM];
    float csum = 0.0f;
    for (int m = 0; m < MM; ++m) {
        float v  = E[(size_t)(n * MM + m) * HH + j];
        float s  = blockReduceSum256(v * v, red);
        float e1 = v / fmaxf(sqrtf(s), 1e-12f);
        float s2 = blockReduceSum256(e1 * e1, red);
        En[(size_t)(n * MM + m) * HH + j] = e1 / fmaxf(sqrtf(s2), 1e-8f);
        ep[m] = e1;
        csum += e1;
    }
    float C  = csum / (float)MM;
    float sc = blockReduceSum256(C * C, red);
    Cn[(size_t)n * HH + j] = C / fmaxf(sqrtf(sc), 1e-8f);
    for (int m = 0; m < MM; ++m) {
        float cm = ((float)MM * C + ep[m]) / (float)(MM - 1);
        float s  = blockReduceSum256(cm * cm, red);
        Cmn[(size_t)(n * MM + m) * HH + j] = cm / fmaxf(sqrtf(s), 1e-8f);
    }
}

__global__ void k_similarity(const float* __restrict__ En, const float* __restrict__ Cn,
                             const float* __restrict__ Cmn,
                             const float* __restrict__ wsim, const float* __restrict__ bsim,
                             float* __restrict__ S) {
    __shared__ float e[HH];
    int b = blockIdx.x, n = threadIdx.x;     // 640 blocks x 64 threads
    for (int k = n; k < HH; k += 64) e[k] = En[(size_t)b * HH + k];
    __syncthreads();
    int own = b / MM;
    const float* cvec = (n == own) ? (Cmn + (size_t)b * HH) : (Cn + (size_t)n * HH);
    float d = 0.0f;
    for (int k = 0; k < HH; ++k) d += e[k] * cvec[k];
    S[(size_t)b * NN + n] = d * (*wsim) + (*bsim);
}

// ---------------- host driver ----------------
extern "C" void kernel_launch(void* const* d_in, const int* in_sizes, int n_in,
                              void* d_out, int out_size, void* d_ws, size_t ws_size,
                              hipStream_t stream) {
    const float* seq  = (const float*)d_in[0];
    const int*   lens = (const int*)  d_in[1];
    const float* wsim = (const float*)d_in[2];
    const float* bsim = (const float*)d_in[3];
    const float* Wih[3] = {(const float*)d_in[4],  (const float*)d_in[8],  (const float*)d_in[12]};
    const float* Whh[3] = {(const float*)d_in[5],  (const float*)d_in[9],  (const float*)d_in[13]};
    const float* bih[3] = {(const float*)d_in[6],  (const float*)d_in[10], (const float*)d_in[14]};
    const float* bhh[3] = {(const float*)d_in[7],  (const float*)d_in[11], (const float*)d_in[15]};

    char* p = (char*)d_ws;
    auto alloc = [&](size_t bytes) -> char* {
        char* r = p; p += (bytes + 255) & ~(size_t)255; return r;
    };
    __bf16* x0   = (__bf16*)alloc((size_t)ROWS * K0P * 2);
    __bf16* hseq = (__bf16*)alloc((size_t)ROWS * HH * 2);
    __bf16* WihB[3]; __bf16* WhhB[3];
    WihB[0] = (__bf16*)alloc((size_t)4 * HH * K0P * 2);
    WihB[1] = (__bf16*)alloc((size_t)4 * HH * HH  * 2);
    WihB[2] = (__bf16*)alloc((size_t)4 * HH * HH  * 2);
    for (int l = 0; l < 3; ++l) WhhB[l] = (__bf16*)alloc((size_t)4 * HH * HH * 2);
    float* bsum  = (float*)alloc(3 * 4 * HH * 4);
    float* clast = (float*)alloc((size_t)BB * HH * 4);
    float* En    = (float*)alloc((size_t)BB * HH * 4);
    float* Cn    = (float*)alloc((size_t)NN * HH * 4);
    float* Cmn   = (float*)alloc((size_t)BB * HH * 4);
    float* S     = (float*)d_out;

    // 1) convert inputs / weights to bf16 (layer0 K padded 40->64)
    k_convert_seq<<<(ROWS * K0P + 255) / 256, 256, 0, stream>>>(seq, x0);
    k_convert_w<<<(4 * HH * K0P + 255) / 256, 256, 0, stream>>>(Wih[0], WihB[0], INF_, K0P);
    k_convert_w<<<(4 * HH * HH + 255) / 256, 256, 0, stream>>>(Wih[1], WihB[1], HH, HH);
    k_convert_w<<<(4 * HH * HH + 255) / 256, 256, 0, stream>>>(Wih[2], WihB[2], HH, HH);
    for (int l = 0; l < 3; ++l) {
        k_convert_w<<<(4 * HH * HH + 255) / 256, 256, 0, stream>>>(Whh[l], WhhB[l], HH, HH);
        k_bias_sum<<<(4 * HH + 255) / 256, 256, 0, stream>>>(bih[l], bhh[l], bsum + l * 4 * HH);
    }

    // 2) three fused LSTM layers (hseq reused in place for layers 1,2)
    k_lstm_layer<K0P><<<BB / 16, 256, 0, stream>>>(x0,   WihB[0], WhhB[0], bsum + 0 * 4 * HH, lens, hseq, clast);
    k_lstm_layer<HH> <<<BB / 16, 256, 0, stream>>>(hseq, WihB[1], WhhB[1], bsum + 1 * 4 * HH, lens, hseq, clast);
    k_lstm_layer<HH> <<<BB / 16, 256, 0, stream>>>(hseq, WihB[2], WhhB[2], bsum + 2 * 4 * HH, lens, hseq, clast);

    // 3) similarity head
    k_embed_norm<<<NN, 256, 0, stream>>>(clast, En, Cn, Cmn);
    k_similarity<<<BB, NN, 0, stream>>>(En, Cn, Cmn, wsim, bsim, S);

    (void)in_sizes; (void)n_in; (void)out_size; (void)ws_size;
}
